// QwenDecoderLayer_27728308862997
// MI455X (gfx1250) — matrix operand order repaired and checked by reading, not compiled, not run on hardware
//
#include <hip/hip_runtime.h>

// ---------------- problem constants ----------------
#define Bb   2
#define Ss   2048
#define Dd   2048
#define NH   16
#define NKV  4
#define HD   128
#define DFF  5632
#define WIN  512

typedef unsigned short u16;
typedef __attribute__((ext_vector_type(16))) __bf16       v16bf;
typedef __attribute__((ext_vector_type(8)))  float        v8f;
typedef __attribute__((ext_vector_type(4)))  unsigned int u32x4;

// ---------------- helpers ----------------
__device__ __forceinline__ u16 f2bf(float f) {
  unsigned u = __float_as_uint(f);
  u += 0x7FFFu + ((u >> 16) & 1u);   // round to nearest even
  return (u16)(u >> 16);
}
__device__ __forceinline__ float bf2f(u16 h) {
  return __uint_as_float(((unsigned)h) << 16);
}

// A-fragment (16x32 bf16, MxK): lane<16 holds row, K 0..15 contiguous; lane>=16 K 16..31
__device__ __forceinline__ v16bf load_fragA(const u16* p0, const u16* p1) {
  union FA { u32x4 u[2]; v16bf v; } x;
  x.u[0] = *(const u32x4*)p0;
  x.u[1] = *(const u32x4*)p1;
  return x.v;
}
// B-fragment (32x16 bf16, KxN): lane<16 holds col, K 0..15 contiguous; lane>=16 K 16..31
__device__ __forceinline__ v16bf load_fragB(const u16* p) {
  union FB { u32x4 u[2]; v16bf v; } x;
  x.u[0] = *(const u32x4*)p;
  x.u[1] = *(const u32x4*)(p + 8);
  return x.v;
}

// ---------------- elementwise kernels ----------------
__global__ __launch_bounds__(256) void cvt_bf16(const float* __restrict__ in,
                                                u16* __restrict__ out, int n) {
  int i = blockIdx.x * 256 + threadIdx.x;
  if (i < n) out[i] = f2bf(in[i]);
}

// one block per row, D = 2048
__global__ __launch_bounds__(256) void rmsnorm_bf16(const float* __restrict__ x,
                                                    const float* __restrict__ w,
                                                    u16* __restrict__ out) {
  __shared__ float red[256];
  const int row = blockIdx.x;
  const float* xr = x + (size_t)row * Dd;
  float ss = 0.f;
  for (int i = threadIdx.x; i < Dd; i += 256) { float v = xr[i]; ss += v * v; }
  red[threadIdx.x] = ss;
  __syncthreads();
  for (int off = 128; off > 0; off >>= 1) {
    if (threadIdx.x < off) red[threadIdx.x] += red[threadIdx.x + off];
    __syncthreads();
  }
  float rs = rsqrtf(red[0] / (float)Dd + 1e-6f);
  for (int i = threadIdx.x; i < Dd; i += 256)
    out[(size_t)row * Dd + i] = f2bf(xr[i] * rs * w[i]);
}

// in: [B,S,H*HD] bf16 ; out: [B,H,S,HD] bf16 with RoPE applied
__global__ __launch_bounds__(256) void rope_permute(const u16* __restrict__ in,
                                                    const float* __restrict__ cosb,
                                                    const float* __restrict__ sinb,
                                                    u16* __restrict__ out, int H) {
  long idx = (long)blockIdx.x * 256 + threadIdx.x;   // out-index order
  int d = idx & (HD - 1);
  long t = idx >> 7;
  int s = t & (Ss - 1);
  t >>= 11;
  int h = (int)(t % H);
  int b = (int)(t / H);
  long ib = ((long)(b * Ss + s)) * (H * HD) + h * HD;
  float xv  = bf2f(in[ib + d]);
  int   d2  = (d < HD / 2) ? d + HD / 2 : d - HD / 2;
  float xr  = bf2f(in[ib + d2]);
  float sgn = (d < HD / 2) ? -1.f : 1.f;
  float c = cosb[s * HD + d], sn = sinb[s * HD + d];
  out[idx] = f2bf(xv * c + sgn * xr * sn);
}

// in: [B,S,NKV*HD] bf16 -> out: [B,NKV,HD,S] bf16 (V transposed for WMMA B-fragments)
__global__ __launch_bounds__(256) void v_transpose(const u16* __restrict__ in,
                                                   u16* __restrict__ out) {
  long idx = (long)blockIdx.x * 256 + threadIdx.x;   // out-index order
  int s = idx & (Ss - 1);
  long t = idx >> 11;
  int d = t & (HD - 1);
  t >>= 7;
  int kv = t & (NKV - 1);
  int b  = (int)(t >> 2);
  out[idx] = in[((long)(b * Ss + s)) * (NKV * HD) + kv * HD + d];
}

__global__ __launch_bounds__(256) void silu_mul(const u16* __restrict__ g,
                                                const u16* __restrict__ u,
                                                u16* __restrict__ out, int n) {
  int i = blockIdx.x * 256 + threadIdx.x;
  if (i < n) {
    float gv = bf2f(g[i]);
    float uv = bf2f(u[i]);
    out[i] = f2bf(gv * uv / (1.f + __expf(-gv)));
  }
}

// ---------------- generic bf16 WMMA GEMM: C[M,N] = A[M,K] * W[N,K]^T ----------------
// Double-buffered LDS with CDNA5 async global->LDS loads (ASYNCcnt pipelined).
// EPI: 1 -> Cf = acc + res ; 2 -> Cb = bf16(acc)
template <int EPI>
__global__ __launch_bounds__(256) void gemm_bf16_wmma(
    const u16* __restrict__ A, const u16* __restrict__ W, int M, int N, int K,
    float* __restrict__ Cf, u16* __restrict__ Cb, const float* __restrict__ res) {
  __shared__ alignas(16) u16 As[2][128 * 32];
  __shared__ alignas(16) u16 Bs[2][128 * 32];

  const int tid  = threadIdx.x;
  const int wave = tid >> 5, lane = tid & 31;
  const int wm = wave >> 1, wn = wave & 1;     // 4x2 wave grid -> 128x128 block tile
  const int hf = lane >> 4, ln = lane & 15;
  const int m0 = blockIdx.y * 128, n0 = blockIdx.x * 128;

  // LDS byte offsets (generic-pointer low 32 bits = LDS offset)
  const unsigned ldsA = (unsigned)(unsigned long long)(void*)&As[0][0];
  const unsigned ldsB = (unsigned)(unsigned long long)(void*)&Bs[0][0];

  // per-thread copy slots: 2 x 16B chunks of A tile + 2 of B tile per k-step
  const int r0 = tid >> 2;         // rows r0 and r0+64
  const int k8 = (tid & 3) << 3;   // bf16 element offset within the 32-wide K slab

  // issue one 8KB+8KB tile pair as 4 async b128 loads straight into LDS
  auto issue_tile = [&](int k0, int buf) {
#pragma unroll
    for (int r = 0; r < 2; ++r) {
      int row = r0 + r * 64;
      unsigned la = ldsA + (unsigned)((buf * 4096 + row * 32 + k8) * 2);
      unsigned long long ga =
          (unsigned long long)(A + (size_t)(m0 + row) * K + k0 + k8);
      asm volatile("global_load_async_to_lds_b128 %0, %1, off"
                   :: "v"(la), "v"(ga) : "memory");
      unsigned lb = ldsB + (unsigned)((buf * 4096 + row * 32 + k8) * 2);
      unsigned long long gb =
          (unsigned long long)(W + (size_t)(n0 + row) * K + k0 + k8);
      asm volatile("global_load_async_to_lds_b128 %0, %1, off"
                   :: "v"(lb), "v"(gb) : "memory");
    }
  };

  v8f acc[2][4];
#pragma unroll
  for (int i = 0; i < 2; i++)
#pragma unroll
    for (int j = 0; j < 4; j++)
#pragma unroll
      for (int e = 0; e < 8; e++) acc[i][j][e] = 0.f;

  issue_tile(0, 0);

  int buf = 0;
  for (int k0 = 0; k0 < K; k0 += 32, buf ^= 1) {
    if (k0 + 32 < K) {
      issue_tile(k0 + 32, buf ^ 1);                 // prefetch next tile
      // async loads drain in order: wait until only the 4 newest remain
      asm volatile("s_wait_asynccnt 4" ::: "memory");
    } else {
      asm volatile("s_wait_asynccnt 0" ::: "memory");
    }
    __syncthreads();                                // all waves' tile-k data in LDS

    const u16* Asb = &As[buf][0];
    const u16* Bsb = &Bs[buf][0];
    v16bf af[2], bfr[4];
#pragma unroll
    for (int i = 0; i < 2; i++) {
      int r = wm * 32 + i * 16 + ln;
      af[i] = load_fragA(&Asb[r * 32 + hf * 16], &Asb[r * 32 + hf * 16 + 8]);
    }
#pragma unroll
    for (int j = 0; j < 4; j++) {
      int r = wn * 64 + j * 16 + ln;
      bfr[j] = load_fragB(&Bsb[r * 32 + hf * 16]);
    }
#pragma unroll
    for (int i = 0; i < 2; i++)
#pragma unroll
      for (int j = 0; j < 4; j++)
        acc[i][j] = __builtin_amdgcn_wmma_f32_16x16x32_bf16(
            false, af[i], false, bfr[j], (short)0, acc[i][j], false, false);
    __syncthreads();   // everyone done reading buf before it is overwritten
  }

#pragma unroll
  for (int i = 0; i < 2; i++)
#pragma unroll
    for (int j = 0; j < 4; j++) {
      int col = n0 + wn * 64 + j * 16 + ln;
#pragma unroll
      for (int g = 0; g < 8; g++) {
        size_t idx = (size_t)(m0 + wm * 32 + i * 16 + hf * 8 + g) * N + col;
        float v = acc[i][j][g];
        if (EPI == 1) Cf[idx] = v + res[idx];
        else          Cb[idx] = f2bf(v);
      }
    }
}

// ---------------- flash-style sliding-window GQA attention ----------------
// Q: [B,NH,S,HD] bf16 ; Kc: [B,NKV,S,HD] bf16 ; Vt: [B,NKV,HD,S] bf16
// O: [B,S,NH*HD] bf16 ; one wave per 16-query tile
__global__ __launch_bounds__(128) void attn_swa(const u16* __restrict__ Q,
                                                const u16* __restrict__ Kc,
                                                const u16* __restrict__ Vt,
                                                u16* __restrict__ O) {
  __shared__ alignas(16) u16 pTile[4][16 * 32];   // per-wave P staging (C-layout -> A-layout)
  const int wave = threadIdx.x >> 5;
  const int lane = threadIdx.x & 31;
  const int hf = lane >> 4, ln = lane & 15;
  const int gtile = blockIdx.x * 4 + wave;
  const int qt = gtile & (Ss / 16 - 1);
  const int h  = (gtile >> 7) & (NH - 1);
  const int b  = gtile >> 11;
  const int q0 = qt * 16;
  const int kvh = h >> 2;                          // NH/NKV = 4

  const u16* Qb = Q  + ((size_t)((b * NH + h) * Ss + q0)) * HD;
  const u16* Kb = Kc + ((size_t)(b * NKV + kvh)) * Ss * HD;
  const u16* Vb = Vt + ((size_t)(b * NKV + kvh)) * HD * Ss;

  // Q fragments (16 queries x 128 hd = 4 A-frags), resident for the whole key loop
  v16bf qf[4];
  {
    const u16* qrow = Qb + ln * HD;
#pragma unroll
    for (int f = 0; f < 4; ++f)
      qf[f] = load_fragA(qrow + f * 32 + hf * 16, qrow + f * 32 + hf * 16 + 8);
  }

  v8f o[8];
#pragma unroll
  for (int j = 0; j < 8; j++)
#pragma unroll
    for (int e = 0; e < 8; e++) o[j][e] = 0.f;
  float mrow[8], lrow[8];
#pragma unroll
  for (int g = 0; g < 8; g++) { mrow[g] = -3.0e38f; lrow[g] = 0.f; }

  int kb0 = q0 - (WIN - 1);
  if (kb0 < 0) kb0 = 0;
  kb0 &= ~31;
  const int kEnd = q0 + 16;
  const float scale = 0.088388347648318447f;       // 1/sqrt(128)

  for (int kk = kb0; kk < kEnd; kk += 32) {
    // ---- S = Q K^T for 32 keys (two 16x16 tiles) ----
    v8f sc[2];
#pragma unroll
    for (int t = 0; t < 2; t++) {
      v8f c;
#pragma unroll
      for (int e = 0; e < 8; e++) c[e] = 0.f;
      const u16* krow = Kb + (size_t)(kk + t * 16 + ln) * HD;
#pragma unroll
      for (int f = 0; f < 4; ++f) {
        v16bf kf = load_fragB(krow + f * 32 + hf * 16);
        c = __builtin_amdgcn_wmma_f32_16x16x32_bf16(false, qf[f], false, kf,
                                                    (short)0, c, false, false);
      }
      sc[t] = c;
    }

    // ---- scale + causal/sliding-window mask (C layout: row = q0+hf*8+g, col = ln) ----
    float rmax[8];
#pragma unroll
    for (int g = 0; g < 8; ++g) {
      int q = q0 + hf * 8 + g;
      int k0i = kk + ln, k1i = kk + 16 + ln;
      float s0 = ((q >= k0i) && (q - k0i < WIN)) ? sc[0][g] * scale : -3.0e38f;
      float s1 = ((q >= k1i) && (q - k1i < WIN)) ? sc[1][g] * scale : -3.0e38f;
      sc[0][g] = s0; sc[1][g] = s1;
      rmax[g] = fmaxf(s0, s1);
    }
#pragma unroll
    for (int msk = 1; msk <= 8; msk <<= 1)
#pragma unroll
      for (int g = 0; g < 8; ++g) rmax[g] = fmaxf(rmax[g], __shfl_xor(rmax[g], msk, 32));

    // ---- online softmax update ----
    float alpha[8], rsum[8];
#pragma unroll
    for (int g = 0; g < 8; ++g) {
      float mn = fmaxf(mrow[g], rmax[g]);
      alpha[g] = __expf(mrow[g] - mn);
      mrow[g] = mn;
      float p0 = (sc[0][g] > -1.0e37f) ? __expf(sc[0][g] - mn) : 0.f;
      float p1 = (sc[1][g] > -1.0e37f) ? __expf(sc[1][g] - mn) : 0.f;
      sc[0][g] = p0; sc[1][g] = p1;
      rsum[g] = p0 + p1;
    }
#pragma unroll
    for (int msk = 1; msk <= 8; msk <<= 1)
#pragma unroll
      for (int g = 0; g < 8; ++g) rsum[g] += __shfl_xor(rsum[g], msk, 32);
#pragma unroll
    for (int g = 0; g < 8; ++g) lrow[g] = lrow[g] * alpha[g] + rsum[g];
#pragma unroll
    for (int j = 0; j < 8; ++j)
#pragma unroll
      for (int g = 0; g < 8; ++g) o[j][g] *= alpha[g];

    // ---- P (C layout) -> LDS -> A fragment; per-wave slice, no block sync needed ----
    u16* pt = pTile[wave];
#pragma unroll
    for (int t = 0; t < 2; t++)
#pragma unroll
      for (int g = 0; g < 8; ++g)
        pt[(hf * 8 + g) * 32 + t * 16 + ln] = f2bf(sc[t][g]);

    v16bf pf = load_fragA(&pt[ln * 32 + hf * 16], &pt[ln * 32 + hf * 16 + 8]);

    // ---- O += P * V  (8 hd tiles of 16; Vt rows are key-contiguous) ----
#pragma unroll
    for (int j = 0; j < 8; ++j) {
      const u16* vrow = Vb + (size_t)(j * 16 + ln) * Ss + kk;
      v16bf vf = load_fragB(vrow + hf * 16);
      o[j] = __builtin_amdgcn_wmma_f32_16x16x32_bf16(false, pf, false, vf,
                                                     (short)0, o[j], false, false);
    }
  }

  // ---- normalize and store [B,S,NH*HD] ----
#pragma unroll
  for (int j = 0; j < 8; ++j)
#pragma unroll
    for (int g = 0; g < 8; ++g) {
      float v = o[j][g] / lrow[g];
      O[(size_t)(b * Ss + q0 + hf * 8 + g) * (NH * HD) + h * HD + j * 16 + ln] = f2bf(v);
    }
}

// ---------------- launcher ----------------
extern "C" void kernel_launch(void* const* d_in, const int* in_sizes, int n_in,
                              void* d_out, int out_size, void* d_ws, size_t ws_size,
                              hipStream_t stream) {
  const float* x    = (const float*)d_in[0];
  const float* cosb = (const float*)d_in[1];
  const float* sinb = (const float*)d_in[2];
  const float* ln1w = (const float*)d_in[3];
  const float* ln2w = (const float*)d_in[4];
  const float* wq   = (const float*)d_in[5];
  const float* wk   = (const float*)d_in[6];
  const float* wv   = (const float*)d_in[7];
  const float* wo   = (const float*)d_in[8];
  const float* wg   = (const float*)d_in[9];
  const float* wu   = (const float*)d_in[10];
  const float* wd   = (const float*)d_in[11];
  float* out = (float*)d_out;

  const int MT = Bb * Ss;  // 4096 token rows

  char* base = (char*)d_ws;
  size_t off = 0;
  auto alloc = [&](size_t elems, size_t esz) -> void* {
    void* p = base + off;
    off += (elems * esz + 255) & ~(size_t)255;
    return p;
  };

  u16* Wq = (u16*)alloc((size_t)Dd * Dd, 2);
  u16* Wk = (u16*)alloc((size_t)NKV * HD * Dd, 2);
  u16* Wv = (u16*)alloc((size_t)NKV * HD * Dd, 2);
  u16* Wo = (u16*)alloc((size_t)Dd * Dd, 2);
  u16* Wg = (u16*)alloc((size_t)DFF * Dd, 2);
  u16* Wu = (u16*)alloc((size_t)DFF * Dd, 2);
  u16* Wd = (u16*)alloc((size_t)Dd * DFF, 2);
  u16* h1 = (u16*)alloc((size_t)MT * Dd, 2);
  u16* qb = (u16*)alloc((size_t)MT * NH * HD, 2);
  u16* kb = (u16*)alloc((size_t)MT * NKV * HD, 2);
  u16* vb = (u16*)alloc((size_t)MT * NKV * HD, 2);
  u16* qr = (u16*)alloc((size_t)MT * NH * HD, 2);
  u16* kr = (u16*)alloc((size_t)MT * NKV * HD, 2);
  u16* vt = (u16*)alloc((size_t)MT * NKV * HD, 2);
  u16* ao = (u16*)alloc((size_t)MT * NH * HD, 2);
  float* x1 = (float*)alloc((size_t)MT * Dd, 4);
  u16* h2 = (u16*)alloc((size_t)MT * Dd, 2);
  u16* gb = (u16*)alloc((size_t)MT * DFF, 2);
  u16* ub = (u16*)alloc((size_t)MT * DFF, 2);
  u16* gu = (u16*)alloc((size_t)MT * DFF, 2);

  // 1) weights -> bf16
  auto cv = [&](const float* s, u16* d, int n) {
    cvt_bf16<<<(n + 255) / 256, 256, 0, stream>>>(s, d, n);
  };
  cv(wq, Wq, Dd * Dd);
  cv(wk, Wk, NKV * HD * Dd);
  cv(wv, Wv, NKV * HD * Dd);
  cv(wo, Wo, Dd * Dd);
  cv(wg, Wg, DFF * Dd);
  cv(wu, Wu, DFF * Dd);
  cv(wd, Wd, Dd * DFF);

  // 2) h1 = rmsnorm(x, ln1_w)   (bf16)
  rmsnorm_bf16<<<MT, 256, 0, stream>>>(x, ln1w, h1);

  // 3) QKV projections (WMMA)
  gemm_bf16_wmma<2><<<dim3((NH * HD) / 128, MT / 128), 256, 0, stream>>>(
      h1, Wq, MT, NH * HD, Dd, nullptr, qb, nullptr);
  gemm_bf16_wmma<2><<<dim3((NKV * HD) / 128, MT / 128), 256, 0, stream>>>(
      h1, Wk, MT, NKV * HD, Dd, nullptr, kb, nullptr);
  gemm_bf16_wmma<2><<<dim3((NKV * HD) / 128, MT / 128), 256, 0, stream>>>(
      h1, Wv, MT, NKV * HD, Dd, nullptr, vb, nullptr);

  // 4) RoPE + layout permutes
  rope_permute<<<(Bb * NH * Ss * HD) / 256, 256, 0, stream>>>(qb, cosb, sinb, qr, NH);
  rope_permute<<<(Bb * NKV * Ss * HD) / 256, 256, 0, stream>>>(kb, cosb, sinb, kr, NKV);
  v_transpose<<<(Bb * NKV * HD * Ss) / 256, 256, 0, stream>>>(vb, vt);

  // 5) sliding-window GQA attention (WMMA)
  attn_swa<<<(Bb * NH * (Ss / 16)) / 4, 128, 0, stream>>>(qr, kr, vt, ao);

  // 6) x1 = x + attn_out @ Wo^T (WMMA + residual)
  gemm_bf16_wmma<1><<<dim3(Dd / 128, MT / 128), 256, 0, stream>>>(
      ao, Wo, MT, Dd, NH * HD, x1, nullptr, x);

  // 7) h2 = rmsnorm(x1, ln2_w)
  rmsnorm_bf16<<<MT, 256, 0, stream>>>(x1, ln2w, h2);

  // 8) gate/up projections (WMMA)
  gemm_bf16_wmma<2><<<dim3(DFF / 128, MT / 128), 256, 0, stream>>>(
      h2, Wg, MT, DFF, Dd, nullptr, gb, nullptr);
  gemm_bf16_wmma<2><<<dim3(DFF / 128, MT / 128), 256, 0, stream>>>(
      h2, Wu, MT, DFF, Dd, nullptr, ub, nullptr);

  // 9) gu = silu(gate) * up
  silu_mul<<<(MT * DFF) / 256, 256, 0, stream>>>(gb, ub, gu, MT * DFF);

  // 10) out = x1 + gu @ Wd^T (WMMA + residual)
  gemm_bf16_wmma<1><<<dim3(Dd / 128, MT / 128), 256, 0, stream>>>(
      gu, Wd, MT, Dd, DFF, out, nullptr, x1);
}
